// BiLSTM_Parser_75823352643746
// MI455X (gfx1250) — matrix operand order repaired and checked by reading, not compiled, not run on hardware
//
#include <hip/hip_runtime.h>
#include <hip/hip_bf16.h>

// ---------------------------------------------------------------------------
// BiLSTM parser for MI455X (gfx1250, wave32).
//   - bf16 WMMA (v_wmma_f32_16x16x32_bf16) for all batched GEMMs, f32 accum
//   - LDS-resident sequential LSTM scan, 1 workgroup per direction
//   - async global->LDS double-buffering of the next timestep's gate row
//     (GLOBAL_LOAD_ASYNC_TO_LDS_B32 + s_wait_asynccnt) to hide the 4KB fetch
//     behind the recurrent GEMV
// Assumed input flattening (setup_inputs dict order, recursive):
//   0 word_tensor 1 pos_tensor 2 word_emb 3 pos_emb
//   4..11  L0: Wih_f Whh_f bih_f bhh_f Wih_b Whh_b bih_b bhh_b
//   12..19 L1: same
//   20 Wh 21 bh 22 Wm 23 bm 24 Wo 25 bo
// ---------------------------------------------------------------------------

typedef __attribute__((ext_vector_type(16))) __bf16 v16bf;
typedef __attribute__((ext_vector_type(8)))  __bf16 v8bf;
typedef __attribute__((ext_vector_type(8)))  float  v8f;

#define T_LEN 1024
#define HID   256      // per-direction hidden
#define G4H   1024     // 4*HID gates

// Async copy of one f32 from global to LDS (ASYNCcnt-tracked, per-lane).
// LDS dest VGPR takes the workgroup-relative LDS byte offset, which is the
// low 32 bits of the flat shared-memory pointer (ISA 10.2 LDS aperture).
__device__ __forceinline__ void async_f32_to_lds(const float* gptr, float* lptr) {
    unsigned loff = (unsigned)(uintptr_t)lptr;
    asm volatile("global_load_async_to_lds_b32 %0, %1, off"
                 :: "v"(loff), "v"(gptr) : "memory");
}
__device__ __forceinline__ void wait_asynccnt0() {
    asm volatile("s_wait_asynccnt 0x0" ::: "memory");
}

// ---------------------------------------------------------------------------
// Embedding gather -> bf16 x0 [1024][320]
// ---------------------------------------------------------------------------
__global__ void embed_gather_kernel(const int* __restrict__ word,
                                    const int* __restrict__ pos,
                                    const float* __restrict__ wemb,
                                    const float* __restrict__ pemb,
                                    __bf16* __restrict__ x0) {
    int t = blockIdx.x;
    int j = threadIdx.x;          // 0..319
    float v;
    if (j < 256) v = wemb[(size_t)word[t] * 256 + j];
    else         v = pemb[(size_t)pos[t] * 64 + (j - 256)];
    x0[(size_t)t * 320 + j] = (__bf16)v;
}

// ---------------------------------------------------------------------------
// f32 -> bf16 elementwise convert
// ---------------------------------------------------------------------------
__global__ void cvt_bf16_kernel(const float* __restrict__ src,
                                __bf16* __restrict__ dst, int n) {
    int i = blockIdx.x * 256 + threadIdx.x;
    if (i < n) dst[i] = (__bf16)src[i];
}

// ---------------------------------------------------------------------------
// Whh [1024][256] -> WhhT [256][1024] (coalesced writes)
// ---------------------------------------------------------------------------
__global__ void transpose_whh_kernel(const float* __restrict__ src,
                                     float* __restrict__ dst) {
    int id = blockIdx.x * 256 + threadIdx.x;   // 0..262143
    int k = id >> 10;          // 0..255
    int n = id & 1023;         // 0..1023
    dst[id] = src[(size_t)n * 256 + k];
}

// ---------------------------------------------------------------------------
// WMMA GEMM:  C[M][N] (f32, row-major) = A[M][K](bf16) * W[N][K](bf16)^T
//             + bias1[N] (+ bias2[N])
// One wave per 16x16 output tile, 4 waves per block. K % 32 == 0.
// Fragment layouts per CDNA5 ISA 7.12.2 (wave32):
//   A lane l (l16=l&15, hi=l>>4): row tm+l16; elems 0..7 = K[k0+hi*8 ..],
//                                 elems 8..15 = K[k0+16+hi*8 ..]
//   B lane l: col tn+l16; elems 0..7 = K[k0+hi*8 ..], elems 8..15 = K[k0+16+hi*8 ..]
//   C lane l, vgpr r: (M = tm + hi*8 + r, N = tn + l16)
// ---------------------------------------------------------------------------
__global__ void wmma_gemm_bias_kernel(const __bf16* __restrict__ A,
                                      const __bf16* __restrict__ W,
                                      const float* __restrict__ bias1,
                                      const float* __restrict__ bias2,
                                      float* __restrict__ C,
                                      int M, int N, int K) {
    int lane = threadIdx.x & 31;
    int wv   = threadIdx.x >> 5;
    int tile = blockIdx.x * 4 + wv;
    int ntN  = N >> 4;
    int tm   = (tile / ntN) << 4;
    int tn   = (tile % ntN) << 4;
    int l16  = lane & 15;
    int hi   = lane >> 4;

    float bv = 0.0f;
    if (bias1) bv += bias1[tn + l16];
    if (bias2) bv += bias2[tn + l16];
    v8f acc = {bv, bv, bv, bv, bv, bv, bv, bv};

    const __bf16* ap = A + (size_t)(tm + l16) * K;
    const __bf16* wp = W + (size_t)(tn + l16) * K;

    for (int k0 = 0; k0 < K; k0 += 32) {
        v8bf alo = *(const v8bf*)(ap + k0 + hi * 8);
        v8bf ahi = *(const v8bf*)(ap + k0 + 16 + hi * 8);
        v16bf av = __builtin_shufflevector(alo, ahi,
                     0, 1, 2, 3, 4, 5, 6, 7, 8, 9, 10, 11, 12, 13, 14, 15);
        v8bf wlo = *(const v8bf*)(wp + k0 + hi * 8);
        v8bf whi = *(const v8bf*)(wp + k0 + 16 + hi * 8);
        v16bf wvv = __builtin_shufflevector(wlo, whi,
                     0, 1, 2, 3, 4, 5, 6, 7, 8, 9, 10, 11, 12, 13, 14, 15);
        acc = __builtin_amdgcn_wmma_f32_16x16x32_bf16(
                  false, av, false, wvv, (short)0, acc, false, false);
    }

    float* cp = C + (size_t)(tm + hi * 8) * N + tn + l16;
#pragma unroll
    for (int r = 0; r < 8; ++r) cp[(size_t)r * N] = acc[r];
}

// ---------------------------------------------------------------------------
// Sequential bidirectional LSTM scan. grid=2 (dir), block=1024 (32 waves).
// xg: precomputed x*Wih^T + (bih+bhh), [1024][1024] f32.
// whhT: [256][1024] f32 (L2 resident, coalesced across the workgroup).
// x_next: [1024][512] f32, fwd -> cols 0..255, bwd -> cols 256..511.
// Gate order (torch): i, f, g, o.
// The gate row for step t+1 is fetched into LDS asynchronously while the
// GEMV for step t runs; thread j touches only element j of the staging
// buffers, so a per-wave s_wait_asynccnt 0 (no barrier) is sufficient.
// ---------------------------------------------------------------------------
__global__ void lstm_scan_kernel(const float* __restrict__ xg_f,
                                 const float* __restrict__ xg_b,
                                 const float* __restrict__ whhT_f,
                                 const float* __restrict__ whhT_b,
                                 float* __restrict__ x_next) {
    __shared__ __align__(16) float h_s[HID];
    __shared__ __align__(16) float c_s[HID];
    __shared__ __align__(16) float g_s[G4H];
    __shared__ __align__(16) float xg_stage[2][G4H];

    const int dir = blockIdx.x;
    const float* __restrict__ xg   = dir ? xg_b   : xg_f;
    const float* __restrict__ whhT = dir ? whhT_b : whhT_f;
    const int j = threadIdx.x;

    if (j < HID) { h_s[j] = 0.0f; c_s[j] = 0.0f; }

    // Kick off the async fetch of step 0's gate row.
    {
        const int t0 = dir ? (T_LEN - 1) : 0;
        async_f32_to_lds(&xg[(size_t)t0 * G4H + j], &xg_stage[0][j]);
    }
    __syncthreads();

    for (int s = 0; s < T_LEN; ++s) {
        const int t   = dir ? (T_LEN - 1 - s) : s;
        const int cur = s & 1;

        // Wait for this step's staged gate row, consume it, then immediately
        // start the async fetch of the next row into the other buffer.
        wait_asynccnt0();
        float acc = xg_stage[cur][j];
        if (s + 1 < T_LEN) {
            const int tn = dir ? (t - 1) : (t + 1);
            async_f32_to_lds(&xg[(size_t)tn * G4H + j], &xg_stage[cur ^ 1][j]);
        }

        // GEMV: g_j = xg[t][j] + sum_k h[k] * Whh[j][k]   (whhT[k][j])
#pragma unroll 4
        for (int k = 0; k < HID; k += 4) {
            float4 hv = *(const float4*)&h_s[k];   // LDS broadcast (uniform addr)
            acc += hv.x * whhT[(size_t)(k + 0) * G4H + j];
            acc += hv.y * whhT[(size_t)(k + 1) * G4H + j];
            acc += hv.z * whhT[(size_t)(k + 2) * G4H + j];
            acc += hv.w * whhT[(size_t)(k + 3) * G4H + j];
        }

        // Activation: tanh on the cell-candidate block [512,768), sigmoid else.
        float act;
        if (j >= 512 && j < 768) act = tanhf(acc);
        else                     act = 1.0f / (1.0f + __expf(-acc));
        g_s[j] = act;
        __syncthreads();

        if (j < HID) {
            float ig = g_s[j];
            float fg = g_s[256 + j];
            float gg = g_s[512 + j];
            float og = g_s[768 + j];
            float c  = fg * c_s[j] + ig * gg;
            float hh = og * tanhf(c);
            c_s[j] = c;
            h_s[j] = hh;
            x_next[(size_t)t * 512 + dir * HID + j] = hh;
        }
        __syncthreads();
    }
}

// ---------------------------------------------------------------------------
// out[row] = sum_j tanh(hm[row][j]) * wo[j]   (hm [1024][256])
// ---------------------------------------------------------------------------
__global__ void tanh_dot_kernel(const float* __restrict__ hm,
                                const float* __restrict__ wo,
                                float* __restrict__ out) {
    __shared__ float red[256];
    int row = blockIdx.x, tid = threadIdx.x;
    red[tid] = tanhf(hm[(size_t)row * 256 + tid]) * wo[tid];
    __syncthreads();
    for (int stp = 128; stp > 0; stp >>= 1) {
        if (tid < stp) red[tid] += red[tid + stp];
        __syncthreads();
    }
    if (tid == 0) out[row] = red[0];
}

// ---------------------------------------------------------------------------
// scores[h][m] = a[h] + b[m] + bo; NaN on diagonal.
// ---------------------------------------------------------------------------
__global__ void scores_kernel(const float* __restrict__ a,
                              const float* __restrict__ b,
                              const float* __restrict__ bo,
                              float* __restrict__ out) {
    int id = blockIdx.x * 256 + threadIdx.x;     // 0 .. 1M-1
    int hr = id >> 10;
    int m  = id & 1023;
    float v = a[hr] + b[m] + bo[0];
    out[id] = (hr == m) ? __uint_as_float(0x7fc00000u) : v;
}

// ---------------------------------------------------------------------------
// Host-side orchestration
// ---------------------------------------------------------------------------
extern "C" void kernel_launch(void* const* d_in, const int* in_sizes, int n_in,
                              void* d_out, int out_size, void* d_ws, size_t ws_size,
                              hipStream_t stream) {
    (void)in_sizes; (void)n_in; (void)out_size; (void)ws_size;

    const int*   word   = (const int*)d_in[0];
    const int*   pos    = (const int*)d_in[1];
    const float* wemb   = (const float*)d_in[2];
    const float* pemb   = (const float*)d_in[3];
    // LSTM params: layer L base = 4 + L*8 ; order Wih_f Whh_f bih_f bhh_f Wih_b Whh_b bih_b bhh_b
    const float* Wh  = (const float*)d_in[20];
    const float* bh  = (const float*)d_in[21];
    const float* Wm  = (const float*)d_in[22];
    const float* bm  = (const float*)d_in[23];
    const float* Wo  = (const float*)d_in[24];
    const float* bo  = (const float*)d_in[25];
    float* out = (float*)d_out;

    // Workspace layout (bytes)
    uint8_t* w = (uint8_t*)d_ws;
    float*  xg_f   = (float*)(w + (size_t)0);              // 4 MB  [1024][1024]
    float*  xg_b   = (float*)(w + ((size_t)4  << 20));     // 4 MB
    float*  whhT_f = (float*)(w + ((size_t)8  << 20));     // 1 MB  [256][1024]
    float*  whhT_b = (float*)(w + ((size_t)9  << 20));     // 1 MB
    float*  x_next = (float*)(w + ((size_t)10 << 20));     // 2 MB  [1024][512]
    __bf16* xA     = (__bf16*)(w + ((size_t)12 << 20));    // 1 MB  activations bf16
    __bf16* wB     = (__bf16*)(w + ((size_t)13 << 20));    // 1 MB  weights bf16
    float*  hm     = (float*)(w + ((size_t)14 << 20));     // 1 MB  [1024][256]
    float*  avec   = (float*)(w + ((size_t)15 << 20));     // 4 KB
    float*  bvec   = (float*)(w + ((size_t)15 << 20) + 4096);

    // 1) Embedding gather -> bf16 x0 [1024][320]
    embed_gather_kernel<<<T_LEN, 320, 0, stream>>>(word, pos, wemb, pemb, xA);

    // 2) Two bidirectional LSTM layers
    int K = 320;
    for (int L = 0; L < 2; ++L) {
        const int base = 4 + L * 8;
        const float* Wih_f = (const float*)d_in[base + 0];
        const float* Whh_f = (const float*)d_in[base + 1];
        const float* bih_f = (const float*)d_in[base + 2];
        const float* bhh_f = (const float*)d_in[base + 3];
        const float* Wih_b = (const float*)d_in[base + 4];
        const float* Whh_b = (const float*)d_in[base + 5];
        const float* bih_b = (const float*)d_in[base + 6];
        const float* bhh_b = (const float*)d_in[base + 7];

        const int nWih = G4H * K;
        const int gemmBlocks = (T_LEN / 16) * (G4H / 16) / 4;  // 4 waves/block

        // fwd input GEMM:  xg_f = x * Wih_f^T + (bih_f + bhh_f)
        cvt_bf16_kernel<<<(nWih + 255) / 256, 256, 0, stream>>>(Wih_f, wB, nWih);
        wmma_gemm_bias_kernel<<<gemmBlocks, 128, 0, stream>>>(
            xA, wB, bih_f, bhh_f, xg_f, T_LEN, G4H, K);

        // bwd input GEMM (stream-serialized, safe to reuse wB)
        cvt_bf16_kernel<<<(nWih + 255) / 256, 256, 0, stream>>>(Wih_b, wB, nWih);
        wmma_gemm_bias_kernel<<<gemmBlocks, 128, 0, stream>>>(
            xA, wB, bih_b, bhh_b, xg_b, T_LEN, G4H, K);

        // Whh transposes for coalesced GEMV in the scan
        transpose_whh_kernel<<<(HID * G4H) / 256, 256, 0, stream>>>(Whh_f, whhT_f);
        transpose_whh_kernel<<<(HID * G4H) / 256, 256, 0, stream>>>(Whh_b, whhT_b);

        // Sequential scan: both directions concurrently (grid=2)
        lstm_scan_kernel<<<2, 1024, 0, stream>>>(xg_f, xg_b, whhT_f, whhT_b, x_next);

        // bf16 copy of layer output for the next GEMM
        const int nX = T_LEN * 512;
        cvt_bf16_kernel<<<(nX + 255) / 256, 256, 0, stream>>>(x_next, xA, nX);
        K = 512;
    }

    // 3) Head / modifier projections (WMMA) + tanh-dot reductions
    const int nW = 256 * 512;
    const int projBlocks = (T_LEN / 16) * (256 / 16) / 4;

    cvt_bf16_kernel<<<(nW + 255) / 256, 256, 0, stream>>>(Wh, wB, nW);
    wmma_gemm_bias_kernel<<<projBlocks, 128, 0, stream>>>(
        xA, wB, bh, nullptr, hm, T_LEN, 256, 512);
    tanh_dot_kernel<<<T_LEN, 256, 0, stream>>>(hm, Wo, avec);          // wo[:256]

    cvt_bf16_kernel<<<(nW + 255) / 256, 256, 0, stream>>>(Wm, wB, nW);
    wmma_gemm_bias_kernel<<<projBlocks, 128, 0, stream>>>(
        xA, wB, bm, nullptr, hm, T_LEN, 256, 512);
    tanh_dot_kernel<<<T_LEN, 256, 0, stream>>>(hm, Wo + 256, bvec);    // wo[256:]

    // 4) Pairwise scores with NaN diagonal
    scores_kernel<<<(T_LEN * T_LEN) / 256, 256, 0, stream>>>(avec, bvec, bo, out);
}
